// MultiHeadAttention_34909494182596
// MI455X (gfx1250) — matrix operand order repaired and checked by reading, not compiled, not run on hardware
//
#include <hip/hip_runtime.h>

#define BATCH 4
#define HEADS 16
#define SEQ   2048
#define MODEL 1024
#define HDIM  64

typedef __attribute__((ext_vector_type(16))) __bf16 v16bf;
typedef __attribute__((ext_vector_type(8)))  __bf16 v8bf;
typedef __attribute__((ext_vector_type(8)))  float  v8f;
typedef __attribute__((ext_vector_type(4)))  float  v4f;

static __device__ __forceinline__ v8f wmma_bf16(v16bf a, v16bf b, v8f c) {
  // D = A(16x32 bf16) * B(32x16 bf16) + C(16x16 f32)
  return __builtin_amdgcn_wmma_f32_16x16x32_bf16(false, a, false, b, (short)0, c,
                                                 false, false);
}

static __device__ __forceinline__ v16bf cat8(v8bf lo, v8bf hi) {
  return __builtin_shufflevector(lo, hi, 0,1,2,3,4,5,6,7,8,9,10,11,12,13,14,15);
}

// ---------------------------------------------------------------- fp32 -> bf16
// 8 elements per thread per iteration: 2x b128 loads -> 1x b128 store.
__global__ __launch_bounds__(256) void mha_cvt_bf16(const float* __restrict__ src,
                                                    __bf16* __restrict__ dst, int n) {
  int i = (blockIdx.x * blockDim.x + threadIdx.x) * 8;
  int stride = gridDim.x * blockDim.x * 8;
  for (; i < n; i += stride) {
    v4f f0 = *(const v4f*)(src + i);
    v4f f1 = *(const v4f*)(src + i + 4);
    v8bf o;
#pragma unroll
    for (int j = 0; j < 4; ++j) { o[j] = (__bf16)f0[j]; o[4 + j] = (__bf16)f1[j]; }
    *(v8bf*)(dst + i) = o;
  }
}

// ------------------------------------------------- projection GEMM: X @ W^T
// Per-wave tile: 32 rows x 64 cols. Two A fragments share four staged B frags.
// mode 0: store bf16 to out[b][h][s][d]   (Q, K heads)
// mode 1: store bf16 to out[b][h][d][s]   (V transposed for P@V B-fragments)
__global__ __launch_bounds__(256) void mha_proj(const __bf16* __restrict__ X,
                                                const __bf16* __restrict__ W,
                                                __bf16* __restrict__ out, int mode) {
  const int wid  = (blockIdx.x * blockDim.x + threadIdx.x) >> 5;
  const int lane = threadIdx.x & 31;
  const int hf   = lane >> 4;
  const int l16  = lane & 15;
  const int m0 = (wid >> 4) * 32;   // 256 row tiles of 32
  const int n0 = (wid & 15) * 64;   // 16 col strips of 64

  const __bf16* arow0 = X + (size_t)(m0 + l16) * MODEL;
  const __bf16* arow1 = X + (size_t)(m0 + 16 + l16) * MODEL;
  v8f c[2][4];
#pragma unroll
  for (int t = 0; t < 2; ++t)
#pragma unroll
    for (int g = 0; g < 4; ++g)
#pragma unroll
      for (int j = 0; j < 8; ++j) c[t][g][j] = 0.0f;

  for (int kb = 0; kb < MODEL; kb += 32) {
    __builtin_prefetch((const void*)(arow0 + kb + 64), 0, 3);
    __builtin_prefetch((const void*)(arow1 + kb + 64), 0, 3);
    v16bf a0 = cat8(*(const v8bf*)(arow0 + kb + hf * 16),
                    *(const v8bf*)(arow0 + kb + 8 + hf * 16));
    v16bf a1 = cat8(*(const v8bf*)(arow1 + kb + hf * 16),
                    *(const v8bf*)(arow1 + kb + 8 + hf * 16));
    v16bf bfr[4];
#pragma unroll
    for (int g = 0; g < 4; ++g)
      bfr[g] = *(const v16bf*)(W + (size_t)(n0 + g * 16 + l16) * MODEL + kb + hf * 16);
#pragma unroll
    for (int g = 0; g < 4; ++g) c[0][g] = wmma_bf16(a0, bfr[g], c[0][g]);
#pragma unroll
    for (int g = 0; g < 4; ++g) c[1][g] = wmma_bf16(a1, bfr[g], c[1][g]);
  }

  const int b  = m0 / SEQ;
  const int s0 = m0 - b * SEQ;
#pragma unroll
  for (int t = 0; t < 2; ++t) {
    if (mode == 0) {
#pragma unroll
      for (int g = 0; g < 4; ++g) {
        int n = n0 + g * 16 + l16;
        int hh = n >> 6, d = n & 63;
        __bf16* dst = out + (((size_t)b * HEADS + hh) * SEQ) * HDIM + d;
#pragma unroll
        for (int r = 0; r < 8; ++r) {
          int sl = s0 + t * 16 + hf * 8 + r;
          dst[(size_t)sl * HDIM] = (__bf16)c[t][g][r];
        }
      }
    } else {
#pragma unroll
      for (int g = 0; g < 4; ++g) {
        int n = n0 + g * 16 + l16;
        int hh = n >> 6, d = n & 63;
        int sl = s0 + t * 16 + hf * 8;  // 8 consecutive s values per lane
        __bf16* dst = out + (((size_t)b * HEADS + hh) * HDIM + d) * SEQ + sl;
        v8bf vv;
#pragma unroll
        for (int r = 0; r < 8; ++r) vv[r] = (__bf16)c[t][g][r];
        *(v8bf*)dst = vv;
      }
    }
  }
}

// ----------------------------- scores: raw S = Q@K^T/8 + mask, online max/sum
__global__ __launch_bounds__(256) void mha_scores(const __bf16* __restrict__ Qh,
                                                  const __bf16* __restrict__ Kh,
                                                  const float* __restrict__ mask,
                                                  float* __restrict__ attn,
                                                  float* __restrict__ rowm,
                                                  float* __restrict__ rowl) {
  const int wid  = (blockIdx.x * blockDim.x + threadIdx.x) >> 5;
  const int lane = threadIdx.x & 31;
  const int hf = lane >> 4, l16 = lane & 15;
  const int qt = wid & 127;        // SEQ/16 tiles
  const int bh = wid >> 7;         // b*HEADS + h
  const int b  = bh >> 4;
  const int q0 = qt * 16;

  // Q A-fragments for d=0..31 and d=32..63
  const __bf16* qrow = Qh + ((size_t)bh * SEQ + q0 + l16) * HDIM;
  v16bf A0 = cat8(*(const v8bf*)(qrow + hf * 16),      *(const v8bf*)(qrow + 8 + hf * 16));
  v16bf A1 = cat8(*(const v8bf*)(qrow + 32 + hf * 16), *(const v8bf*)(qrow + 40 + hf * 16));

  float mrun[8], lrun[8];
#pragma unroll
  for (int r = 0; r < 8; ++r) { mrun[r] = -1e30f; lrun[r] = 0.0f; }

  const __bf16* kbase = Kh + (size_t)bh * SEQ * HDIM;
  float* arow0 = attn + ((size_t)bh * SEQ + q0) * SEQ;
  const float* maskb = mask + (size_t)b * SEQ;

  for (int kt = 0; kt < SEQ / 16; ++kt) {
    const __bf16* krow = kbase + (size_t)(kt * 16 + l16) * HDIM + hf * 16;
    __builtin_prefetch((const void*)(krow + 16 * HDIM), 0, 3);  // next key tile
    v16bf B0 = *(const v16bf*)krow;
    v16bf B1 = *(const v16bf*)(krow + 32);
    v8f c;
#pragma unroll
    for (int j = 0; j < 8; ++j) c[j] = 0.0f;
    c = wmma_bf16(A0, B0, c);
    c = wmma_bf16(A1, B1, c);

    float mv = maskb[kt * 16 + l16];        // column n = l16 of this tile
#pragma unroll
    for (int r = 0; r < 8; ++r) c[r] = c[r] * 0.125f + mv;

    // store raw scores: row m = r + 8*hf, col = kt*16 + l16
#pragma unroll
    for (int r = 0; r < 8; ++r)
      arow0[(size_t)(hf * 8 + r) * SEQ + kt * 16 + l16] = c[r];

    // online row max / sum-of-exp across the 16 columns (lanes of our half)
#pragma unroll
    for (int r = 0; r < 8; ++r) {
      float t = c[r];
      t = fmaxf(t, __shfl_xor(t, 1)); t = fmaxf(t, __shfl_xor(t, 2));
      t = fmaxf(t, __shfl_xor(t, 4)); t = fmaxf(t, __shfl_xor(t, 8));
      float nm = fmaxf(mrun[r], t);
      float e = __expf(c[r] - nm);
      e += __shfl_xor(e, 1); e += __shfl_xor(e, 2);
      e += __shfl_xor(e, 4); e += __shfl_xor(e, 8);
      lrun[r] = lrun[r] * __expf(mrun[r] - nm) + e;
      mrun[r] = nm;
    }
  }

  if (l16 == 0) {  // lane 0 -> rows 0..7, lane 16 -> rows 8..15
    float* pm = rowm + (size_t)bh * SEQ + q0 + hf * 8;
    float* pl = rowl + (size_t)bh * SEQ + q0 + hf * 8;
#pragma unroll
    for (int r = 0; r < 8; ++r) { pm[r] = mrun[r]; pl[r] = lrun[r]; }
  }
}

// ------------------- pass 2: normalize attn in place, ctx = P @ V (via Vt)
__global__ __launch_bounds__(256) void mha_pv(float* __restrict__ attn,
                                              const __bf16* __restrict__ Vt,
                                              const float* __restrict__ rowm,
                                              const float* __restrict__ rowl,
                                              __bf16* __restrict__ ctx) {
  const int wid  = (blockIdx.x * blockDim.x + threadIdx.x) >> 5;
  const int lane = threadIdx.x & 31;
  const int hf = lane >> 4, l16 = lane & 15;
  const int qt = wid & 127;
  const int bh = wid >> 7;
  const int b = bh >> 4, h = bh & 15;
  const int q0 = qt * 16;
  const int row = q0 + l16;                 // A-fragment: lane row = l16

  const float m    = rowm[(size_t)bh * SEQ + row];
  const float linv = 1.0f / rowl[(size_t)bh * SEQ + row];
  float* arow = attn + ((size_t)bh * SEQ + row) * SEQ;
  const __bf16* vtb = Vt + (size_t)bh * HDIM * SEQ;

  v8f c[4];
#pragma unroll
  for (int g = 0; g < 4; ++g)
#pragma unroll
    for (int j = 0; j < 8; ++j) c[g][j] = 0.0f;

  for (int kb = 0; kb < SEQ; kb += 32) {
    // stage the four V B-fragments first so loads overlap the exp math below
    v16bf vf[4];
#pragma unroll
    for (int g = 0; g < 4; ++g)
      vf[g] = *(const v16bf*)(vtb + (size_t)(g * 16 + l16) * SEQ + kb + hf * 16);

    float* p0 = arow + kb + hf * 16;      // k = hf*16 + 0..7
    float* p1 = arow + kb + 8 + hf * 16;  // k = hf*16 + 8..15
    v4f f0 = *(const v4f*)p0, f1 = *(const v4f*)(p0 + 4);
    v4f f2 = *(const v4f*)p1, f3 = *(const v4f*)(p1 + 4);
#pragma unroll
    for (int j = 0; j < 4; ++j) {
      f0[j] = __expf(f0[j] - m) * linv;
      f1[j] = __expf(f1[j] - m) * linv;
      f2[j] = __expf(f2[j] - m) * linv;
      f3[j] = __expf(f3[j] - m) * linv;
    }
    *(v4f*)p0 = f0; *(v4f*)(p0 + 4) = f1;   // normalized attn output
    *(v4f*)p1 = f2; *(v4f*)(p1 + 4) = f3;

    v16bf a;
#pragma unroll
    for (int j = 0; j < 4; ++j) {
      a[j]      = (__bf16)f0[j];
      a[4 + j]  = (__bf16)f1[j];
      a[8 + j]  = (__bf16)f2[j];
      a[12 + j] = (__bf16)f3[j];
    }
#pragma unroll
    for (int g = 0; g < 4; ++g) c[g] = wmma_bf16(a, vf[g], c[g]);
  }

  // ctx[b][s][h*64+d], C-tile: lane col d = g*16+l16, rows m = r + 8*hf
#pragma unroll
  for (int g = 0; g < 4; ++g) {
    int d = g * 16 + l16;
#pragma unroll
    for (int r = 0; r < 8; ++r) {
      int sl = q0 + hf * 8 + r;
      ctx[((size_t)b * SEQ + sl) * MODEL + h * HDIM + d] = (__bf16)c[g][r];
    }
  }
}

// ------------------------------------------------- final fc: out = ctx @ Wfc^T
__global__ __launch_bounds__(256) void mha_fc(const __bf16* __restrict__ X,
                                              const __bf16* __restrict__ W,
                                              float* __restrict__ out) {
  const int wid  = (blockIdx.x * blockDim.x + threadIdx.x) >> 5;
  const int lane = threadIdx.x & 31;
  const int hf = lane >> 4, l16 = lane & 15;
  const int m0 = (wid >> 4) * 32;
  const int n0 = (wid & 15) * 64;

  const __bf16* arow0 = X + (size_t)(m0 + l16) * MODEL;
  const __bf16* arow1 = X + (size_t)(m0 + 16 + l16) * MODEL;
  v8f c[2][4];
#pragma unroll
  for (int t = 0; t < 2; ++t)
#pragma unroll
    for (int g = 0; g < 4; ++g)
#pragma unroll
      for (int j = 0; j < 8; ++j) c[t][g][j] = 0.0f;

  for (int kb = 0; kb < MODEL; kb += 32) {
    __builtin_prefetch((const void*)(arow0 + kb + 64), 0, 3);
    __builtin_prefetch((const void*)(arow1 + kb + 64), 0, 3);
    v16bf a0 = cat8(*(const v8bf*)(arow0 + kb + hf * 16),
                    *(const v8bf*)(arow0 + kb + 8 + hf * 16));
    v16bf a1 = cat8(*(const v8bf*)(arow1 + kb + hf * 16),
                    *(const v8bf*)(arow1 + kb + 8 + hf * 16));
    v16bf bfr[4];
#pragma unroll
    for (int g = 0; g < 4; ++g)
      bfr[g] = *(const v16bf*)(W + (size_t)(n0 + g * 16 + l16) * MODEL + kb + hf * 16);
#pragma unroll
    for (int g = 0; g < 4; ++g) c[0][g] = wmma_bf16(a0, bfr[g], c[0][g]);
#pragma unroll
    for (int g = 0; g < 4; ++g) c[1][g] = wmma_bf16(a1, bfr[g], c[1][g]);
  }

#pragma unroll
  for (int t = 0; t < 2; ++t)
#pragma unroll
    for (int g = 0; g < 4; ++g) {
      int n = n0 + g * 16 + l16;
#pragma unroll
      for (int r = 0; r < 8; ++r) {
        int mrow = m0 + t * 16 + hf * 8 + r;
        out[(size_t)mrow * MODEL + n] = c[t][g][r];
      }
    }
}

extern "C" void kernel_launch(void* const* d_in, const int* in_sizes, int n_in,
                              void* d_out, int out_size, void* d_ws, size_t ws_size,
                              hipStream_t stream) {
  (void)in_sizes; (void)n_in; (void)out_size; (void)ws_size;
  const float* q    = (const float*)d_in[0];
  const float* k    = (const float*)d_in[1];
  const float* v    = (const float*)d_in[2];
  const float* mask = (const float*)d_in[3];
  const float* w_qs = (const float*)d_in[4];
  const float* w_ks = (const float*)d_in[5];
  const float* w_vs = (const float*)d_in[6];
  const float* w_fc = (const float*)d_in[7];

  float* out  = (float*)d_out;                      // [4,2048,1024]
  float* attn = out + (size_t)BATCH * SEQ * MODEL;  // [4,16,2048,2048]

  char* ws = (char*)d_ws;
  const size_t MB = 1024 * 1024;
  __bf16* qb  = (__bf16*)(ws + 0 * MB);
  __bf16* kb  = (__bf16*)(ws + 16 * MB);
  __bf16* vb  = (__bf16*)(ws + 32 * MB);
  __bf16* wqb = (__bf16*)(ws + 48 * MB);
  __bf16* wkb = (__bf16*)(ws + 50 * MB);
  __bf16* wvb = (__bf16*)(ws + 52 * MB);
  __bf16* wfb = (__bf16*)(ws + 54 * MB);
  __bf16* Qh  = (__bf16*)(ws + 56 * MB);   // [B,H,S,64]
  __bf16* Kh  = (__bf16*)(ws + 72 * MB);   // [B,H,S,64]
  __bf16* Vt  = (__bf16*)(ws + 88 * MB);   // [B,H,64,S]
  float*  rowm = (float*)(ws + 104 * MB);  // [B,H,S]
  float*  rowl = (float*)(ws + 105 * MB);  // [B,H,S]
  __bf16* ctx = (__bf16*)(ws + 106 * MB);  // [B,S,MODEL]

  const int nAct = BATCH * SEQ * MODEL;    // 8388608
  const int nW   = MODEL * MODEL;          // 1048576

  mha_cvt_bf16<<<1024, 256, 0, stream>>>(q, qb, nAct);
  mha_cvt_bf16<<<1024, 256, 0, stream>>>(k, kb, nAct);
  mha_cvt_bf16<<<1024, 256, 0, stream>>>(v, vb, nAct);
  mha_cvt_bf16<<<128, 256, 0, stream>>>(w_qs, wqb, nW);
  mha_cvt_bf16<<<128, 256, 0, stream>>>(w_ks, wkb, nW);
  mha_cvt_bf16<<<128, 256, 0, stream>>>(w_vs, wvb, nW);
  mha_cvt_bf16<<<128, 256, 0, stream>>>(w_fc, wfb, nW);

  // GEMMs: 4096 waves = 512 blocks * 8 wave32 (32x64 tiles)
  mha_proj<<<512, 256, 0, stream>>>(qb, wqb, Qh, 0);
  mha_proj<<<512, 256, 0, stream>>>(kb, wkb, Kh, 0);
  mha_proj<<<512, 256, 0, stream>>>(vb, wvb, Vt, 1);

  // attention: 8192 waves = 1024 blocks * 8 wave32 (16-query tiles)
  mha_scores<<<1024, 256, 0, stream>>>(Qh, Kh, mask, attn, rowm, rowl);
  mha_pv<<<1024, 256, 0, stream>>>(attn, Vt, rowm, rowl, ctx);
  mha_fc<<<512, 256, 0, stream>>>(ctx, wfb, out);
}
